// ResidualVectorQuantization_19215683682940
// MI455X (gfx1250) — matrix operand order repaired and checked by reading, not compiled, not run on hardware
//
#include <hip/hip_runtime.h>
#include <stdint.h>

typedef __attribute__((ext_vector_type(16))) __bf16 v16bf;
typedef __attribute__((ext_vector_type(8)))  float  v8f;
typedef __attribute__((ext_vector_type(4)))  unsigned int u32x4;
typedef __attribute__((ext_vector_type(8)))  int          i32x8;
typedef __attribute__((ext_vector_type(4)))  int          i32x4;

constexpr int Bn = 8, Dn = 256, Tn = 4096, NQn = 8, Kn = 1024;
constexpr int BTn = Bn * Tn;   // 32768 tokens
constexpr int CHK  = 16;       // codes staged per chunk
constexpr int NCHK = Kn / CHK; // 64 chunks
constexpr int ROWP = Dn + 16;  // LDS row pitch in bf16 elems: 512B + 32B pad

#ifndef USE_TDM
#  if __has_builtin(__builtin_amdgcn_tensor_load_to_lds) && \
      __has_builtin(__builtin_amdgcn_s_wait_tensorcnt)
#    define USE_TDM 1
#  else
#    define USE_TDM 0
#  endif
#endif

// ---------------------------------------------------------------------------
// round-to-nearest-even fp32 -> bf16 pair pack
__device__ __forceinline__ uint32_t pack2_bf16(float a, float b) {
  uint32_t ua = __float_as_uint(a), ub = __float_as_uint(b);
  ua = (ua + 0x7FFFu + ((ua >> 16) & 1u)) >> 16;
  ub = (ub + 0x7FFFu + ((ub >> 16) & 1u)) >> 16;
  return (ua & 0xFFFFu) | (ub << 16);
}

__device__ __forceinline__ uint32_t lds_off(const void* p) {
  return (uint32_t)(uintptr_t)p;   // LDS aperture: addr[31:0] is the LDS offset
}

#if USE_TDM
// Issue a TDM load of a 16-row x 256-col bf16 tile into LDS with 32B row pad.
// D# per CDNA5 ISA (group0 128b, group1 256b, groups2/3 = 0).
// 6-arg builtin form (clang-23 / therock-10 lane).
__device__ __forceinline__ void tdm_issue(const uint16_t* gsrc, uint32_t ldsbyte) {
  const uint64_t ga = (uint64_t)(uintptr_t)gsrc;
  const u32x4 g0 = {
      1u,                                               // count=1, user desc
      ldsbyte,                                          // lds_addr
      (uint32_t)ga,                                     // global_addr[31:0]
      ((uint32_t)(ga >> 32) & 0x01FFFFFFu) | (2u << 30) // gaddr[56:32], type=2
  };
  const i32x8 g1 = {
      (int)((1u << 16) |      // data_size = 2 bytes
            (1u << 20) |      // pad_enable
            (6u << 22) |      // pad_interval: 128 DWORDs (one 512B row)
            (7u << 25)),      // pad_amount: 8 DWORDs (32B)
      (int)(256u << 16),      // tensor_dim0[15:0] = 256 (bits 63:48)
      (int)(1024u << 16),     // tensor_dim1[15:0] = 1024 (bits 95:80)
      (int)(256u << 16),      // tile_dim0 = 256 (bits 127:112)
      16,                     // tile_dim1 = 16 (bits 143:128), tile_dim2 = 0
      256,                    // tensor_dim0_stride[31:0] = 256
      0, 0                    // dim0_stride hi, dim1_stride (unused, 2-D tile)
  };
  const i32x4 gz4 = {0, 0, 0, 0};
  const i32x8 gz8 = {0, 0, 0, 0, 0, 0, 0, 0};
  __builtin_amdgcn_tensor_load_to_lds(g0, g1, gz4, gz4, gz8, 0);
}
#endif

// ---------------------------------------------------------------------------
// init: transpose x [B,D,T] -> residual [B,T,D]; zero qout; zero loss sums
__global__ __launch_bounds__(256) void k_init(const float* __restrict__ x,
                                              float* __restrict__ residual,
                                              float* __restrict__ qout,
                                              float* __restrict__ loss_sum) {
  const int e = blockIdx.x * 256 + threadIdx.x;          // < B*T*D
  const int b = e / (Tn * Dn);
  const int rem = e - b * (Tn * Dn);
  const int t = rem / Dn;
  const int d = rem - t * Dn;
  residual[e] = x[((size_t)b * Dn + d) * Tn + t];
  qout[e] = 0.0f;
  if (e < NQn) loss_sum[e] = 0.0f;
}

// ---------------------------------------------------------------------------
// codebook prep: fp32 [NQ*K, D] -> bf16, plus row squared norms
__global__ __launch_bounds__(256) void k_prep_cb(const float* __restrict__ cb,
                                                 uint16_t* __restrict__ cb_bf,
                                                 float* __restrict__ c2) {
  __shared__ float red[256];
  const int rowi = blockIdx.x;                           // [0, NQ*K)
  const int d = threadIdx.x;
  const float v = cb[(size_t)rowi * Dn + d];
  const uint32_t u = __float_as_uint(v);
  cb_bf[(size_t)rowi * Dn + d] =
      (uint16_t)((u + 0x7FFFu + ((u >> 16) & 1u)) >> 16);
  red[d] = v * v;
  __syncthreads();
  for (int s = 128; s > 0; s >>= 1) {
    if (d < s) red[d] += red[d + s];
    __syncthreads();
  }
  if (d == 0) c2[rowi] = red[0];
}

// ---------------------------------------------------------------------------
// WMMA distance + argmin. Block = 8 waves x 16 tokens = 128 tokens.
// Codebook chunks are staged in LDS (TDM double-buffered) and shared by all
// waves. dist surrogate = ||c||^2 - 2*(r.c), dot via v_wmma_f32_16x16x32_bf16.
__global__ __launch_bounds__(256) void k_argmin(
    const float* __restrict__ residual,      // [BT, D] fp32
    const uint16_t* __restrict__ cb_bf,      // [K, D] bf16 (this stage)
    const float* __restrict__ c2,            // [K] (this stage)
    int* __restrict__ idx_ws,                // [BT]
    float* __restrict__ idx_out) {           // [BT] as float
  __shared__ __align__(32) uint16_t smB[2][CHK * ROWP];  // 2 x 8.5 KB
  __shared__ float smC2[Kn];                             // 4 KB

  const int tid  = threadIdx.x;
  const int lane = tid & 31;
  const int wave = tid >> 5;
  const int half = lane >> 4;                // 0: rows 0-7 / K-lo, 1: rows 8-15 / K-hi
  const int lr   = lane & 15;
  const int rowbase = (blockIdx.x * 8 + wave) * 16;
  const int row = rowbase + lr;

#if USE_TDM
  if (wave == 0) tdm_issue(cb_bf, lds_off(&smB[0][0]));  // prefetch chunk 0
#endif

  // stage ||c||^2 into LDS
  for (int i = tid; i < Kn; i += 256) smC2[i] = c2[i];

  // Build all 8 A-fragments (D = 256 = 8 * 32) once; keep in registers.
  // A 16x32 bf16 layout: lane half h: elems 0..15 = K(h*16+0..15)
  union AU { v16bf v; uint32_t u[8]; };
  AU afrag[8];
  const float* rp = residual + (size_t)row * Dn;
#pragma unroll
  for (int s = 0; s < 8; ++s) {
    const int k0 = s * 32 + half * 16;
    const float4 f0 = *reinterpret_cast<const float4*>(rp + k0);
    const float4 f1 = *reinterpret_cast<const float4*>(rp + k0 + 4);
    const float4 f2 = *reinterpret_cast<const float4*>(rp + k0 + 8);
    const float4 f3 = *reinterpret_cast<const float4*>(rp + k0 + 12);
    afrag[s].u[0] = pack2_bf16(f0.x, f0.y);
    afrag[s].u[1] = pack2_bf16(f0.z, f0.w);
    afrag[s].u[2] = pack2_bf16(f1.x, f1.y);
    afrag[s].u[3] = pack2_bf16(f1.z, f1.w);
    afrag[s].u[4] = pack2_bf16(f2.x, f2.y);
    afrag[s].u[5] = pack2_bf16(f2.z, f2.w);
    afrag[s].u[6] = pack2_bf16(f3.x, f3.y);
    afrag[s].u[7] = pack2_bf16(f3.z, f3.w);
  }

  float bestv[8];
  int   besti[8];
#pragma unroll
  for (int v = 0; v < 8; ++v) { bestv[v] = 3.4e38f; besti[v] = 0; }

#if USE_TDM
  if (wave == 0) __builtin_amdgcn_s_wait_tensorcnt(0);   // chunk 0 landed
  __syncthreads();                                       // publish to block
#endif

  for (int c = 0; c < NCHK; ++c) {
#if USE_TDM
    const int cur = c & 1;
    if (c + 1 < NCHK && wave == 0)                       // overlap DMA w/ WMMA
      tdm_issue(cb_bf + (size_t)(c + 1) * CHK * Dn, lds_off(&smB[1 - cur][0]));
#else
    const int cur = 0;
    __syncthreads();
    {  // cooperative copy of chunk c: 256 threads x 32B
      const int r = tid >> 4, sgm = tid & 15;
      const uint4* src = reinterpret_cast<const uint4*>(
          cb_bf + ((size_t)c * CHK + r) * Dn + sgm * 16);
      uint4* dst = reinterpret_cast<uint4*>(&smB[0][r * ROWP + sgm * 16]);
      dst[0] = src[0];
      dst[1] = src[1];
    }
    __syncthreads();
#endif
    const int n = c * CHK + lr;              // this lane's candidate code
    v8f acc = {};
    // B 32x16 bf16: lane col = lr, elems 0..15 = K(half*16 + 0..15); 32B in LDS
    const uint16_t* bb = &smB[cur][lr * ROWP + half * 16];
#pragma unroll
    for (int s = 0; s < 8; ++s) {
      const v16bf b = *reinterpret_cast<const v16bf*>(bb + s * 32);
      acc = __builtin_amdgcn_wmma_f32_16x16x32_bf16(
          false, afrag[s].v, false, b, (short)0, acc, false, false);
    }
    const float c2n = smC2[n];
#pragma unroll
    for (int v = 0; v < 8; ++v) {
      const float dv = c2n - 2.0f * acc[v];  // row = rowbase + half*8 + v
      if (dv < bestv[v] || (dv == bestv[v] && n < besti[v])) {
        bestv[v] = dv;
        besti[v] = n;
      }
    }
#if USE_TDM
    if (c + 1 < NCHK && wave == 0) __builtin_amdgcn_s_wait_tensorcnt(0);
    __syncthreads();                         // next buffer ready / reads done
#endif
  }

  // min-reduce across the 16 lanes sharing each row group (xor stays in-group)
#pragma unroll
  for (int m = 1; m <= 8; m <<= 1) {
#pragma unroll
    for (int v = 0; v < 8; ++v) {
      const float ov = __shfl_xor(bestv[v], m, 32);
      const int   oi = __shfl_xor(besti[v], m, 32);
      if (ov < bestv[v] || (ov == bestv[v] && oi < besti[v])) {
        bestv[v] = ov;
        besti[v] = oi;
      }
    }
  }

  if (lr == 0) {
#pragma unroll
    for (int v = 0; v < 8; ++v) {
      const int tok = rowbase + half * 8 + v;
      idx_ws[tok]  = besti[v];
      idx_out[tok] = (float)besti[v];
    }
  }
}

// ---------------------------------------------------------------------------
// gather q (exact fp32 codebook), update residual & qout, accumulate loss
__global__ __launch_bounds__(256) void k_update(
    const float* __restrict__ cb,            // [K, D] fp32 (this stage)
    const int* __restrict__ idx_ws,
    float* __restrict__ residual,
    float* __restrict__ qout,
    float* __restrict__ loss_sum) {          // scalar for this stage
  __shared__ float red[256];
  const int tok = blockIdx.x;
  const int d = threadIdx.x;
  const int idx = idx_ws[tok];
  const float q = cb[(size_t)idx * Dn + d];
  const float r = residual[(size_t)tok * Dn + d];
  const float diff = q - r;
  residual[(size_t)tok * Dn + d] = r - q;
  qout[(size_t)tok * Dn + d] += q;
  red[d] = diff * diff;
  __syncthreads();
  for (int s = 128; s > 0; s >>= 1) {
    if (d < s) red[d] += red[d + s];
    __syncthreads();
  }
  if (d == 0) atomicAdd(loss_sum, red[0]);
}

// ---------------------------------------------------------------------------
__global__ void k_losses(const float* __restrict__ loss_sum,
                         float* __restrict__ out) {
  const int i = threadIdx.x;
  if (i < NQn) out[i] = 2.0f * loss_sum[i] / (float)((size_t)BTn * Dn);
}

// ---------------------------------------------------------------------------
extern "C" void kernel_launch(void* const* d_in, const int* in_sizes, int n_in,
                              void* d_out, int out_size, void* d_ws,
                              size_t ws_size, hipStream_t stream) {
  const float* x  = (const float*)d_in[0];   // [B, D, T]
  const float* cb = (const float*)d_in[1];   // [NQ, K, D]

  float* out        = (float*)d_out;
  float* qout       = out;                                  // B*T*D
  float* idx_out    = out + (size_t)BTn * Dn;               // NQ*B*T (as float)
  float* losses_out = idx_out + (size_t)NQn * BTn;          // NQ

  char* ws = (char*)d_ws;
  float*    residual = (float*)ws;    ws += (size_t)BTn * Dn * sizeof(float);
  uint16_t* cb_bf    = (uint16_t*)ws; ws += (size_t)NQn * Kn * Dn * sizeof(uint16_t);
  float*    c2       = (float*)ws;    ws += (size_t)NQn * Kn * sizeof(float);
  int*      idx_ws   = (int*)ws;      ws += (size_t)BTn * sizeof(int);
  float*    loss_sum = (float*)ws;

  k_init<<<(BTn * Dn) / 256, 256, 0, stream>>>(x, residual, qout, loss_sum);
  k_prep_cb<<<NQn * Kn, 256, 0, stream>>>(cb, cb_bf, c2);

  for (int n = 0; n < NQn; ++n) {
    k_argmin<<<BTn / 128, 256, 0, stream>>>(
        residual, cb_bf + (size_t)n * Kn * Dn, c2 + (size_t)n * Kn,
        idx_ws, idx_out + (size_t)n * BTn);
    k_update<<<BTn, 256, 0, stream>>>(
        cb + (size_t)n * Kn * Dn, idx_ws, residual, qout, loss_sum + n);
  }

  k_losses<<<1, 32, 0, stream>>>(loss_sum, losses_out);
}
